// GroupedQueryAttention_66022237274135
// MI455X (gfx1250) — matrix operand order repaired and checked by reading, not compiled, not run on hardware
//
#include <hip/hip_runtime.h>

// ---------------- problem constants ----------------
#define D_MODEL 2048
#define NHEADS  32
#define NKV     8
#define HD      64
#define SEQ     2048
#define BATCH   2
#define MROWS   (BATCH*SEQ)      // 4096
#define KVDIM   (NKV*HD)         // 512

typedef unsigned short u16;
typedef __attribute__((ext_vector_type(16))) __bf16 v16bf;
typedef __attribute__((ext_vector_type(8)))  float  v8f;
typedef __attribute__((ext_vector_type(8)))  u16    u16x8;
typedef __attribute__((ext_vector_type(2)))  unsigned int u32x2;
typedef int i32x4 __attribute__((vector_size(16)));   // matches builtin param type

#define AS1 __attribute__((address_space(1)))
#define AS3 __attribute__((address_space(3)))

#if __has_builtin(__builtin_amdgcn_global_load_async_to_lds_b128)
#define HAVE_ASYNC_LDS 1
#endif
#if __has_builtin(__builtin_amdgcn_cvt_pk_bf16_f32)
#define HAVE_PK_BF16 1
#endif
#if __has_builtin(__builtin_amdgcn_s_wait_asynccnt)
#define WAIT_ASYNC() __builtin_amdgcn_s_wait_asynccnt(0)
#else
#define WAIT_ASYNC() asm volatile("s_wait_asynccnt 0" ::: "memory")
#endif

union FragBF {            // one WMMA 16-bit operand (16 bf16 per lane = 8 VGPRs)
  v16bf bf;
  u16x8 u[2];
};

__device__ __forceinline__ u16 f2bf(float f) {   // RNE f32->bf16
  unsigned int u = __float_as_uint(f);
  u += 0x7fffu + ((u >> 16) & 1u);
  return (u16)(u >> 16);
}

__device__ __forceinline__ unsigned int f2bf2(float lo, float hi) { // packed 2xbf16
#ifdef HAVE_PK_BF16
  auto r = __builtin_amdgcn_cvt_pk_bf16_f32(lo, hi);
  union { decltype(r) v; unsigned int u; } cv;
  cv.v = r;
  return cv.u;
#else
  return (unsigned int)f2bf(lo) | ((unsigned int)f2bf(hi) << 16);
#endif
}

// LDS row strides (u16 units): padded for bank spread, 16B-aligned rows
#define LAS 40
#define LBS 40
#define LQS 72
#define LKS 72
#define LVS 40
#define LPS 40

// =====================================================================
// GEMM: C(MxN,f32) = A(MxK,f32) x B(KxN,f32), bf16 WMMA compute.
// 256 threads = 8 waves; block tile 128x128; BK=32; 8 WMMA/wave/step.
// =====================================================================
__global__ __launch_bounds__(256)
void gqa_gemm_bf16(const float* __restrict__ A, const float* __restrict__ Bw,
                   float* __restrict__ C, int M, int N, int K)
{
  __shared__ u16 lA[128 * LAS];   // [m][k]
  __shared__ u16 lB[128 * LBS];   // [n][k] (W tile transposed)

  const int tid  = threadIdx.x;
  const int wave = tid >> 5;
  const int lane = tid & 31;
  const int hf   = lane >> 4;
  const int ln   = lane & 15;
  const int m0 = blockIdx.y * 128;
  const int n0 = blockIdx.x * 128;
  const int mw = wave & 3;       // owns m-subtiles 2mw, 2mw+1 (rows mw*32..)
  const int nw = wave >> 2;      // owns n-subtiles nw*4 .. nw*4+3

  v8f acc[2][4] = {};

  for (int k0 = 0; k0 < K; k0 += 32) {
    // ---- stage A tile (128x32 f32 -> bf16), coalesced float4 reads ----
#pragma unroll
    for (int q = 0; q < 4; ++q) {
      int slot = tid + q * 256;
      int r  = slot >> 3;
      int cg = (slot & 7) << 2;
      float4 f = *(const float4*)(A + (long)(m0 + r) * K + k0 + cg);
      u32x2 p;
      p.x = f2bf2(f.x, f.y);
      p.y = f2bf2(f.z, f.w);
      *(u32x2*)&lA[r * LAS + cg] = p;
    }
    // ---- stage B tile transposed: lane indexes K so LDS writes spread banks;
    //      W stays L2-resident (192MB L2) so per-lane 16B reads are cheap ----
#pragma unroll
    for (int q = 0; q < 4; ++q) {
      int ngb = ((wave + 8 * q) << 2);          // n base 0..124 step 4
      float4 f = *(const float4*)(Bw + (long)(k0 + lane) * N + n0 + ngb);
      unsigned int p0 = f2bf2(f.x, f.y);
      unsigned int p1 = f2bf2(f.z, f.w);
      lB[(ngb + 0) * LBS + lane] = (u16)p0;
      lB[(ngb + 1) * LBS + lane] = (u16)(p0 >> 16);
      lB[(ngb + 2) * LBS + lane] = (u16)p1;
      lB[(ngb + 3) * LBS + lane] = (u16)(p1 >> 16);
    }
    __syncthreads();

    // ---- fragments (CDNA5 16-bit A 16x32 / B 32x16 lane layouts) ----
    FragBF a[2], b[4];
#pragma unroll
    for (int im = 0; im < 2; ++im) {
      int arow = (mw * 32 + im * 16 + ln) * LAS;
      a[im].u[0] = *(const u16x8*)&lA[arow + hf * 8];
      a[im].u[1] = *(const u16x8*)&lA[arow + 16 + hf * 8];
    }
#pragma unroll
    for (int in = 0; in < 4; ++in) {
      int brow = (nw * 64 + in * 16 + ln) * LBS + hf * 8;
      b[in].u[0] = *(const u16x8*)&lB[brow];
      b[in].u[1] = *(const u16x8*)&lB[brow + 16];
    }
#pragma unroll
    for (int im = 0; im < 2; ++im)
#pragma unroll
      for (int in = 0; in < 4; ++in)
        acc[im][in] = __builtin_amdgcn_wmma_f32_16x16x32_bf16(
            false, a[im].bf, false, b[in].bf, (short)0, acc[im][in], false, false);
    __syncthreads();
  }

  // ---- store C ----
#pragma unroll
  for (int im = 0; im < 2; ++im)
#pragma unroll
    for (int in = 0; in < 4; ++in)
#pragma unroll
      for (int g = 0; g < 8; ++g) {
        int row = m0 + mw * 32 + im * 16 + g + 8 * hf;
        int col = n0 + nw * 64 + in * 16 + ln;
        C[(long)row * N + col] = acc[im][in][g];
      }
}

// =====================================================================
// RoPE + permute: pre[(b*S+i)][h*HD+d] f32 -> out[((b*nh+h)*S+i)*HD+d] bf16
// =====================================================================
__global__ __launch_bounds__(256)
void gqa_rope_permute(const float* __restrict__ pre, const float* __restrict__ cosb,
                      const float* __restrict__ sinb, u16* __restrict__ outb, int nh)
{
  long idx = (long)blockIdx.x * 256 + threadIdx.x;
  long total = (long)BATCH * SEQ * nh * 32;
  if (idx >= total) return;
  int dl = idx & 31;
  long t = idx >> 5;
  int h = (int)(t % nh); t /= nh;
  int i = (int)(t % SEQ);
  int b = (int)(t / SEQ);

  long rowoff = (long)(b * SEQ + i) * (nh * HD) + h * HD;
  float x1 = pre[rowoff + dl];
  float x2 = pre[rowoff + dl + 32];
  float c1 = cosb[i * HD + dl],      s1 = sinb[i * HD + dl];
  float c2 = cosb[i * HD + dl + 32], s2 = sinb[i * HD + dl + 32];
  float o1 = x1 * c1 - x2 * s1;      // rotated = [-x2, x1]
  float o2 = x2 * c2 + x1 * s2;

  long obase = ((long)(b * nh + h) * SEQ + i) * HD;
  outb[obase + dl]      = f2bf(o1);
  outb[obase + dl + 32] = f2bf(o2);
}

// V permute+convert: pre[(b*S+j)][g*HD+d] f32 -> out[((b*NKV+g)*S+j)*HD+d] bf16
__global__ __launch_bounds__(256)
void gqa_permute_v(const float* __restrict__ pre, u16* __restrict__ outb)
{
  long idx = (long)blockIdx.x * 256 + threadIdx.x;
  long total = (long)BATCH * SEQ * NKV * HD;
  if (idx >= total) return;
  int d = idx & 63;
  long t = idx >> 6;
  int g = (int)(t & 7); t >>= 3;
  int j = (int)(t & (SEQ - 1));
  int b = (int)(t >> 11);
  float v = pre[(long)(b * SEQ + j) * KVDIM + g * HD + d];
  outb[((long)(b * NKV + g) * SEQ + j) * HD + d] = f2bf(v);
}

// =====================================================================
// Causal flash attention: 128 q rows per block (8 waves x 16 rows),
// key blocks of 32, head_dim 64. q/k staged via async global->LDS copies.
// =====================================================================
__global__ __launch_bounds__(256)
void gqa_attention(const u16* __restrict__ qb, const u16* __restrict__ kb,
                   const u16* __restrict__ vb, float* __restrict__ ctx)
{
  __shared__ u16 lq[128 * LQS];      // q tile [i][d]
  __shared__ u16 lk[32 * LKS];       // k tile [j][d]
  __shared__ u16 lv[64 * LVS];       // v tile transposed [d][j]
  __shared__ u16 lp[8][16 * LPS];    // per-wave P tile [m][j]

  const int tid  = threadIdx.x;
  const int wave = tid >> 5;
  const int lane = tid & 31;
  const int hf   = lane >> 4;
  const int ln   = lane & 15;

  const int bh = blockIdx.y;
  const int b  = bh >> 5;            // bh = b*NHEADS + h
  const int h  = bh & 31;
  const int g  = h >> 2;             // kv group (GROUP_SIZE=4)
  const int i0 = blockIdx.x * 128;

  const u16* qg = qb + ((long)(b * NHEADS + h) * SEQ + i0) * HD;
  const u16* kg = kb + (long)(b * NKV + g) * SEQ * HD;
  const u16* vg = vb + (long)(b * NKV + g) * SEQ * HD;

  // ---- stage q tile (async global->LDS when available) ----
#pragma unroll
  for (int q = 0; q < 4; ++q) {
    int slot = tid + q * 256;                  // 16B chunk id
    int dst  = (slot >> 3) * LQS + (slot & 7) * 8;
#ifdef HAVE_ASYNC_LDS
    __builtin_amdgcn_global_load_async_to_lds_b128(
        (AS1 i32x4*)(qg + slot * 8), (AS3 i32x4*)&lq[dst], 0, 0);
#else
    *(u16x8*)&lq[dst] = *(const u16x8*)&qg[slot * 8];
#endif
  }
#ifdef HAVE_ASYNC_LDS
  WAIT_ASYNC();
#endif
  __syncthreads();

  // persistent q A-fragments for the two d-chunks (d 0..31 / 32..63)
  FragBF aq0, aq1;
  const int qrow = (wave * 16 + ln) * LQS;
  aq0.u[0] = *(const u16x8*)&lq[qrow + hf * 8];
  aq0.u[1] = *(const u16x8*)&lq[qrow + 16 + hf * 8];
  aq1.u[0] = *(const u16x8*)&lq[qrow + 32 + hf * 8];
  aq1.u[1] = *(const u16x8*)&lq[qrow + 48 + hf * 8];

  v8f acc0 = {}, acc1 = {}, acc2 = {}, acc3 = {};
  float mrow[8], lrow[8];
#pragma unroll
  for (int gg = 0; gg < 8; ++gg) { mrow[gg] = -3.0e38f; lrow[gg] = 0.0f; }

  const int nblocks = (blockIdx.x + 1) * 4;   // causal: keys up to i0+127
  for (int jb = 0; jb < nblocks; ++jb) {
    const int j0 = jb * 32;
    // ---- stage k block ----
    {
      int dst = (tid >> 3) * LKS + (tid & 7) * 8;
#ifdef HAVE_ASYNC_LDS
      __builtin_amdgcn_global_load_async_to_lds_b128(
          (AS1 i32x4*)(kg + (long)j0 * 64 + tid * 8), (AS3 i32x4*)&lk[dst], 0, 0);
#else
      *(u16x8*)&lk[dst] = *(const u16x8*)&kg[(long)j0 * 64 + tid * 8];
#endif
    }
    // ---- stage v block transposed: lv[d][j] ----
    {
      int j  = tid >> 3;
      int d8 = (tid & 7) << 3;
      u16x8 vv = *(const u16x8*)&vg[(long)(j0 + j) * 64 + d8];
#pragma unroll
      for (int e = 0; e < 8; ++e) lv[(d8 + e) * LVS + j] = vv[e];
    }
    // prefetch next k/v block while this one is consumed
    if (jb + 1 < nblocks) {
      __builtin_prefetch(kg + (long)(j0 + 32) * 64 + tid * 8, 0, 1);
      __builtin_prefetch(vg + (long)(j0 + 32) * 64 + tid * 8, 0, 1);
    }
#ifdef HAVE_ASYNC_LDS
    WAIT_ASYNC();
#endif
    __syncthreads();

    // ---- scores: S(16q x 32k), K-reduction over d in 2 chunks ----
    v8f s0 = {}, s1 = {};
    FragBF bk;
    int kr = ln * LKS;
    bk.u[0] = *(const u16x8*)&lk[kr + hf * 8];
    bk.u[1] = *(const u16x8*)&lk[kr + 16 + hf * 8];
    s0 = __builtin_amdgcn_wmma_f32_16x16x32_bf16(false, aq0.bf, false, bk.bf, (short)0, s0, false, false);
    bk.u[0] = *(const u16x8*)&lk[kr + 32 + hf * 8];
    bk.u[1] = *(const u16x8*)&lk[kr + 48 + hf * 8];
    s0 = __builtin_amdgcn_wmma_f32_16x16x32_bf16(false, aq1.bf, false, bk.bf, (short)0, s0, false, false);
    kr = (16 + ln) * LKS;
    bk.u[0] = *(const u16x8*)&lk[kr + hf * 8];
    bk.u[1] = *(const u16x8*)&lk[kr + 16 + hf * 8];
    s1 = __builtin_amdgcn_wmma_f32_16x16x32_bf16(false, aq0.bf, false, bk.bf, (short)0, s1, false, false);
    bk.u[0] = *(const u16x8*)&lk[kr + 32 + hf * 8];
    bk.u[1] = *(const u16x8*)&lk[kr + 48 + hf * 8];
    s1 = __builtin_amdgcn_wmma_f32_16x16x32_bf16(false, aq1.bf, false, bk.bf, (short)0, s1, false, false);

    // ---- online softmax (scale 1/sqrt(64)=0.125, causal mask) ----
#pragma unroll
    for (int gg = 0; gg < 8; ++gg) {
      int i = i0 + wave * 16 + gg + 8 * hf;
      float x0 = s0[gg] * 0.125f; if (j0 + ln > i)      x0 = -3.0e38f;
      float x1 = s1[gg] * 0.125f; if (j0 + 16 + ln > i) x1 = -3.0e38f;
      float mx = fmaxf(x0, x1);
      mx = fmaxf(mx, __shfl_xor(mx, 1, 32));
      mx = fmaxf(mx, __shfl_xor(mx, 2, 32));
      mx = fmaxf(mx, __shfl_xor(mx, 4, 32));
      mx = fmaxf(mx, __shfl_xor(mx, 8, 32));
      float mnew = fmaxf(mrow[gg], mx);
      float alpha = __expf(mrow[gg] - mnew);
      mrow[gg] = mnew;
      float p0 = __expf(x0 - mnew);
      float p1 = __expf(x1 - mnew);
      float rs = p0 + p1;
      rs += __shfl_xor(rs, 1, 32);
      rs += __shfl_xor(rs, 2, 32);
      rs += __shfl_xor(rs, 4, 32);
      rs += __shfl_xor(rs, 8, 32);
      lrow[gg] = lrow[gg] * alpha + rs;
      acc0[gg] *= alpha; acc1[gg] *= alpha; acc2[gg] *= alpha; acc3[gg] *= alpha;
      int prow = gg + 8 * hf;                 // C-layout -> LDS [m][j]
      lp[wave][prow * LPS + ln]      = f2bf(p0);
      lp[wave][prow * LPS + 16 + ln] = f2bf(p1);
    }
    __syncthreads();

    // ---- context: acc += P(16x32) x V(32x64) ----
    FragBF ap;
    const int pr = ln * LPS;
    ap.u[0] = *(const u16x8*)&lp[wave][pr + hf * 8];
    ap.u[1] = *(const u16x8*)&lp[wave][pr + 16 + hf * 8];
    FragBF bv;
    int vr = ln * LVS + hf * 8;
    bv.u[0] = *(const u16x8*)&lv[vr];      bv.u[1] = *(const u16x8*)&lv[vr + 16];
    acc0 = __builtin_amdgcn_wmma_f32_16x16x32_bf16(false, ap.bf, false, bv.bf, (short)0, acc0, false, false);
    vr = (16 + ln) * LVS + hf * 8;
    bv.u[0] = *(const u16x8*)&lv[vr];      bv.u[1] = *(const u16x8*)&lv[vr + 16];
    acc1 = __builtin_amdgcn_wmma_f32_16x16x32_bf16(false, ap.bf, false, bv.bf, (short)0, acc1, false, false);
    vr = (32 + ln) * LVS + hf * 8;
    bv.u[0] = *(const u16x8*)&lv[vr];      bv.u[1] = *(const u16x8*)&lv[vr + 16];
    acc2 = __builtin_amdgcn_wmma_f32_16x16x32_bf16(false, ap.bf, false, bv.bf, (short)0, acc2, false, false);
    vr = (48 + ln) * LVS + hf * 8;
    bv.u[0] = *(const u16x8*)&lv[vr];      bv.u[1] = *(const u16x8*)&lv[vr + 16];
    acc3 = __builtin_amdgcn_wmma_f32_16x16x32_bf16(false, ap.bf, false, bv.bf, (short)0, acc3, false, false);
    __syncthreads();
  }

  // ---- epilogue: normalize by row-sum, write ctx f32 ----
#pragma unroll
  for (int gg = 0; gg < 8; ++gg) {
    float inv = 1.0f / lrow[gg];
    int i = i0 + wave * 16 + gg + 8 * hf;
    long base = ((long)(b * SEQ + i) * NHEADS + h) * HD;
    ctx[base + ln]      = acc0[gg] * inv;
    ctx[base + 16 + ln] = acc1[gg] * inv;
    ctx[base + 32 + ln] = acc2[gg] * inv;
    ctx[base + 48 + ln] = acc3[gg] * inv;
  }
}

// =====================================================================
extern "C" void kernel_launch(void* const* d_in, const int* in_sizes, int n_in,
                              void* d_out, int out_size, void* d_ws, size_t ws_size,
                              hipStream_t stream)
{
  const float* x    = (const float*)d_in[0];
  const float* Wq   = (const float*)d_in[1];
  const float* Wk   = (const float*)d_in[2];
  const float* Wv   = (const float*)d_in[3];
  const float* Wo   = (const float*)d_in[4];
  const float* cosb = (const float*)d_in[5];
  const float* sinb = (const float*)d_in[6];
  float* out = (float*)d_out;

  char* w = (char*)d_ws;                        // 72 MiB layout
  float* q_pre = (float*)(w);                   // 32 MiB (reused as ctx)
  float* k_pre = (float*)(w + (32l << 20));     //  8 MiB
  float* v_pre = (float*)(w + (40l << 20));     //  8 MiB
  u16*   qbf   = (u16*)  (w + (48l << 20));     // 16 MiB
  u16*   kbf   = (u16*)  (w + (64l << 20));     //  4 MiB
  u16*   vbf   = (u16*)  (w + (68l << 20));     //  4 MiB
  float* ctx   = q_pre;

  dim3 blk(256);
  // QKV projections
  gqa_gemm_bf16<<<dim3(D_MODEL / 128, MROWS / 128), blk, 0, stream>>>(x, Wq, q_pre, MROWS, D_MODEL, D_MODEL);
  gqa_gemm_bf16<<<dim3(KVDIM / 128, MROWS / 128), blk, 0, stream>>>(x, Wk, k_pre, MROWS, KVDIM, D_MODEL);
  gqa_gemm_bf16<<<dim3(KVDIM / 128, MROWS / 128), blk, 0, stream>>>(x, Wv, v_pre, MROWS, KVDIM, D_MODEL);
  // RoPE + layout conversion to bf16
  gqa_rope_permute<<<(BATCH * SEQ * NHEADS * 32) / 256, blk, 0, stream>>>(q_pre, cosb, sinb, qbf, NHEADS);
  gqa_rope_permute<<<(BATCH * SEQ * NKV * 32) / 256, blk, 0, stream>>>(k_pre, cosb, sinb, kbf, NKV);
  gqa_permute_v<<<(BATCH * SEQ * NKV * HD) / 256, blk, 0, stream>>>(v_pre, vbf);
  // flash attention (ctx overwrites q_pre region; reads qbf/kbf/vbf)
  gqa_attention<<<dim3(SEQ / 128, BATCH * NHEADS), blk, 0, stream>>>(qbf, kbf, vbf, ctx);
  // output projection
  gqa_gemm_bf16<<<dim3(D_MODEL / 128, MROWS / 128), blk, 0, stream>>>(ctx, Wo, out, MROWS, D_MODEL, D_MODEL);
}
